// FPModule_68410239091225
// MI455X (gfx1250) — matrix operand order repaired and checked by reading, not compiled, not run on hardware
//
#include <hip/hip_runtime.h>
#include <hip/hip_bf16.h>

// ---------------- Problem constants ----------------
#define N_COARSE 4096
#define N_FINE   16384
#define F_COARSE 512
#define F_SKIP   256
#define F_CAT    768           // F_COARSE + F_SKIP
#define HIDDEN   512
#define OUTF     256
#define N_EDGES  524288
#define KNN_K    3

// ---------------- WMMA vector types ----------------
typedef __attribute__((ext_vector_type(16))) __bf16          v16bf;
typedef __attribute__((ext_vector_type(16))) unsigned short  v16u;
typedef __attribute__((ext_vector_type(8)))  float           v8f;

__device__ __forceinline__ unsigned short f2bf(float f) {
    unsigned int u = __float_as_uint(f);
    unsigned int r = u + 0x7FFFu + ((u >> 16) & 1u);   // round-to-nearest-even
    return (unsigned short)(r >> 16);
}

// ---------------- 1) kNN top-3 (LDS-staged coarse points) ----------------
__global__ __launch_bounds__(256) void knn_kernel(
    const float* __restrict__ pos,        // [N_COARSE,3]
    const int*   __restrict__ batch,      // [N_COARSE]
    const float* __restrict__ pos_skip,   // [N_FINE,3]
    const int*   __restrict__ batch_skip, // [N_FINE]
    int*   __restrict__ idx_out,          // [N_FINE,3]
    float* __restrict__ w_out)            // [N_FINE,3] normalized weights
{
    __shared__ float4 sc[N_COARSE];       // 64 KB of the 320 KB WGP LDS
    for (int i = threadIdx.x; i < N_COARSE; i += 256) {
        sc[i] = make_float4(pos[i*3+0], pos[i*3+1], pos[i*3+2], (float)batch[i]);
    }
    __syncthreads();

    int n = blockIdx.x * 256 + threadIdx.x;
    float px = pos_skip[n*3+0], py = pos_skip[n*3+1], pz = pos_skip[n*3+2];
    float pb = (float)batch_skip[n];

    float d0 = 3.0e38f, d1 = 3.0e38f, d2 = 3.0e38f;
    int   i0 = 0, i1 = 0, i2 = 0;
    for (int j = 0; j < N_COARSE; ++j) {
        float4 c = sc[j];
        float dx = px - c.x, dy = py - c.y, dz = pz - c.z;
        float d = dx*dx + dy*dy + dz*dz;
        if (c.w != pb) d = 1.0e8f;                   // BIG cross-batch mask
        if (d < d2) {
            if (d < d1) {
                d2 = d1; i2 = i1;
                if (d < d0) { d1 = d0; i1 = i0; d0 = d; i0 = j; }
                else        { d1 = d;  i1 = j; }
            } else { d2 = d; i2 = j; }
        }
    }
    float w0 = 1.0f / fmaxf(d0, 1e-16f);
    float w1 = 1.0f / fmaxf(d1, 1e-16f);
    float w2 = 1.0f / fmaxf(d2, 1e-16f);
    float inv = 1.0f / (w0 + w1 + w2);
    idx_out[n*3+0] = i0; idx_out[n*3+1] = i1; idx_out[n*3+2] = i2;
    w_out[n*3+0] = w0*inv; w_out[n*3+1] = w1*inv; w_out[n*3+2] = w2*inv;
}

// ---------------- 2) weight convert + transpose to bf16 [N,K] ----------------
__global__ __launch_bounds__(256) void transpose_tobf16(
    const float* __restrict__ W, unsigned short* __restrict__ Wt, int K, int N)
{
    int t = blockIdx.x * 256 + threadIdx.x;
    if (t >= K * N) return;
    int k = t / N, n = t % N;
    Wt[(long long)n * K + k] = f2bf(W[t]);
}

// ---------------- 3) interpolate + concat -> bf16 Hcat [N_FINE, F_CAT] ----------------
__global__ __launch_bounds__(256) void interp_concat(
    const float* __restrict__ x,       // [N_COARSE, F_COARSE]
    const float* __restrict__ x_skip,  // [N_FINE, F_SKIP]
    const int*   __restrict__ idx,     // [N_FINE,3]
    const float* __restrict__ w,       // [N_FINE,3]
    unsigned short* __restrict__ hcat) // [N_FINE, F_CAT] bf16
{
    int t = blockIdx.x * 256 + threadIdx.x;       // < N_FINE*F_CAT (12.6M)
    int n = t / F_CAT, f = t % F_CAT;
    float v;
    if (f < F_COARSE) {
        int   i0 = idx[n*3+0], i1 = idx[n*3+1], i2 = idx[n*3+2];
        float w0 = w[n*3+0],   w1 = w[n*3+1],   w2 = w[n*3+2];
        v = w0 * x[(long long)i0 * F_COARSE + f]
          + w1 * x[(long long)i1 * F_COARSE + f]
          + w2 * x[(long long)i2 * F_COARSE + f];
    } else {
        v = x_skip[(long long)n * F_SKIP + (f - F_COARSE)];
    }
    hcat[t] = f2bf(v);
}

// -------- 4) WMMA bf16 GEMM, 32x32 C tile per wave (2x2 register blocking) --------
// A row-major bf16 [M,K]; Bt = B transposed, bf16 [N,K]; C f32 [M,N].
// K is a compile-time constant -> exact trip count, no remainder branch.
// Per K-step: 2 A fragments + 2 B fragments -> 4 v_wmma (2x data reuse).

// Load one 16x32 bf16 fragment (ISA 7.12.2 16-bit A layout) from a row base.
__device__ __forceinline__ v16bf load_frag_rowK(const unsigned short* base, int k0, int half) {
    v16u u;
    #pragma unroll
    for (int v = 0; v < 8; ++v) {
        int ka = k0 + (v < 4 ? 2*v : 16 + 2*(v - 4)) + 8 * half;
        unsigned int p = *(const unsigned int*)(base + ka);
        u[2*v]   = (unsigned short)(p & 0xFFFFu);
        u[2*v+1] = (unsigned short)(p >> 16);
    }
    return __builtin_bit_cast(v16bf, u);
}
// Load one 32x16 bf16 B fragment (same K order as the A fragment) from a col base.
__device__ __forceinline__ v16bf load_frag_colK(const unsigned short* base, int k0, int half) {
    v16u u;
    #pragma unroll
    for (int v = 0; v < 8; ++v) {
        int kb = k0 + (v < 4 ? 2*v : 16 + 2*(v - 4)) + 8 * half;
        unsigned int p = *(const unsigned int*)(base + kb);
        u[2*v]   = (unsigned short)(p & 0xFFFFu);
        u[2*v+1] = (unsigned short)(p >> 16);
    }
    return __builtin_bit_cast(v16bf, u);
}

template <int K>
__global__ __launch_bounds__(256) void wmma_gemm_bf16(
    const unsigned short* __restrict__ A,
    const unsigned short* __restrict__ Bt,
    float* __restrict__ C, int M, int N)
{
    int gw   = (blockIdx.x * 256 + threadIdx.x) >> 5;   // global wave id
    int lane = threadIdx.x & 31;
    int tilesN = N >> 5;                                // 32-wide C tiles
    int tm = gw / tilesN;
    int tn = gw % tilesN;
    int half = lane >> 4;          // lane group 0..15 vs 16..31
    int l15  = lane & 15;

    const unsigned short* a0 = A  + (long long)(tm * 32 + l15) * K;
    const unsigned short* a1 = a0 + (long long)16 * K;
    const unsigned short* b0 = Bt + (long long)(tn * 32 + l15) * K;
    const unsigned short* b1 = b0 + (long long)16 * K;

    v8f acc00 = {}, acc01 = {}, acc10 = {}, acc11 = {};
    for (int k0 = 0; k0 < K; k0 += 32) {
        if (k0 + 32 < K) {                      // pull next K-tile toward the WGP
            __builtin_prefetch(a0 + k0 + 32, 0, 1);
            __builtin_prefetch(a1 + k0 + 32, 0, 1);
        }
        v16bf fa0 = load_frag_rowK(a0, k0, half);
        v16bf fa1 = load_frag_rowK(a1, k0, half);
        v16bf fb0 = load_frag_colK(b0, k0, half);
        v16bf fb1 = load_frag_colK(b1, k0, half);
        acc00 = __builtin_amdgcn_wmma_f32_16x16x32_bf16(false, fa0, false, fb0, (short)0, acc00, false, false);
        acc01 = __builtin_amdgcn_wmma_f32_16x16x32_bf16(false, fa0, false, fb1, (short)0, acc01, false, false);
        acc10 = __builtin_amdgcn_wmma_f32_16x16x32_bf16(false, fa1, false, fb0, (short)0, acc10, false, false);
        acc11 = __builtin_amdgcn_wmma_f32_16x16x32_bf16(false, fa1, false, fb1, (short)0, acc11, false, false);
    }
    // C/D layout: group0 rows M=0..7, group1 rows M=8..15; N = l15 (+16 for *1 tiles)
    float* c00 = C + (long long)(tm * 32 + 8 * half) * N + tn * 32 + l15;
    #pragma unroll
    for (int r = 0; r < 8; ++r) {
        long long ro = (long long)r * N;
        c00[ro]                       = acc00[r];
        c00[ro + 16]                  = acc01[r];
        c00[ro + (long long)16 * N]      = acc10[r];
        c00[ro + (long long)16 * N + 16] = acc11[r];
    }
}

// ---------------- 5) GCN normalization helpers ----------------
__global__ __launch_bounds__(256) void deg_init(float* deg) {
    int i = blockIdx.x * 256 + threadIdx.x;
    if (i < N_FINE) deg[i] = 1.0f;                 // self loop
}
__global__ __launch_bounds__(256) void deg_count(const int* __restrict__ ei, float* deg) {
    int e = blockIdx.x * 256 + threadIdx.x;
    if (e < N_EDGES) atomicAdd(&deg[ei[N_EDGES + e]], 1.0f);   // col = target
}
__global__ __launch_bounds__(256) void deg_rsqrt(const float* deg, float* dis) {
    int i = blockIdx.x * 256 + threadIdx.x;
    if (i < N_FINE) dis[i] = rsqrtf(fmaxf(deg[i], 1.0f));
}

// self-loop contribution: out[n,f] = dis[n]^2 * xw[n,f]  (also initializes out)
__global__ __launch_bounds__(256) void self_init(
    const float* __restrict__ xw, const float* __restrict__ dis,
    float* __restrict__ out, int F)
{
    int t = blockIdx.x * 256 + threadIdx.x;
    int n = t / F;
    float d = dis[n];
    out[t] = d * d * xw[t];
}

// edge scatter: one wave per edge, lanes stride the feature dim
__global__ __launch_bounds__(256) void edge_scatter(
    const int* __restrict__ ei, const float* __restrict__ dis,
    const float* __restrict__ xw, float* __restrict__ out, int F)
{
    int wid  = (blockIdx.x * 256 + threadIdx.x) >> 5;
    int lane = threadIdx.x & 31;
    if (wid >= N_EDGES) return;
    int row = ei[wid];
    int col = ei[N_EDGES + wid];
    float nrm = dis[row] * dis[col];
    const float* src = xw  + (long long)row * F;
    float*       dst = out + (long long)col * F;
    for (int f = lane; f < F; f += 32)
        atomicAdd(&dst[f], nrm * src[f]);
}

// bias + relu, optionally emitting bf16 for the next WMMA GEMM
__global__ __launch_bounds__(256) void bias_relu_bf16(
    float* __restrict__ out, const float* __restrict__ b,
    unsigned short* __restrict__ obf, int F)
{
    int t = blockIdx.x * 256 + threadIdx.x;
    float v = fmaxf(out[t] + b[t % F], 0.0f);
    obf[t] = f2bf(v);
}
__global__ __launch_bounds__(256) void bias_relu(
    float* __restrict__ out, const float* __restrict__ b, int F)
{
    int t = blockIdx.x * 256 + threadIdx.x;
    out[t] = fmaxf(out[t] + b[t % F], 0.0f);
}

// ---------------- 6) tuple tail: pos_skip, batch_skip ----------------
__global__ __launch_bounds__(256) void tail_copy(
    const float* __restrict__ pos_skip, const int* __restrict__ batch_skip,
    float* __restrict__ out)
{
    int i = blockIdx.x * 256 + threadIdx.x;
    if (i < N_FINE * 3) out[i] = pos_skip[i];
    if (i < N_FINE)     out[N_FINE * 3 + i] = (float)batch_skip[i];
}

// ---------------- launch ----------------
extern "C" void kernel_launch(void* const* d_in, const int* in_sizes, int n_in,
                              void* d_out, int out_size, void* d_ws, size_t ws_size,
                              hipStream_t stream) {
    const float* x          = (const float*)d_in[0];
    const float* pos        = (const float*)d_in[1];
    const int*   batch      = (const int*)  d_in[2];
    const float* x_skip     = (const float*)d_in[3];
    const float* pos_skip   = (const float*)d_in[4];
    const int*   batch_skip = (const int*)  d_in[5];
    const int*   edge_index = (const int*)  d_in[6];
    const float* W1         = (const float*)d_in[7];
    const float* b1         = (const float*)d_in[8];
    const float* W2         = (const float*)d_in[9];
    const float* b2         = (const float*)d_in[10];
    float* outp = (float*)d_out;

    // linear workspace allocator (256B aligned)
    char* ws = (char*)d_ws;
    size_t off = 0;
    auto alloc = [&](size_t bytes) -> void* {
        void* p = ws + off;
        off += (bytes + 255) & ~(size_t)255;
        return p;
    };
    int*            ws_idx  = (int*)           alloc((size_t)N_FINE * 3 * 4);
    float*          ws_w    = (float*)         alloc((size_t)N_FINE * 3 * 4);
    unsigned short* ws_hcat = (unsigned short*)alloc((size_t)N_FINE * F_CAT * 2);
    unsigned short* ws_w1t  = (unsigned short*)alloc((size_t)HIDDEN * F_CAT * 2);
    unsigned short* ws_w2t  = (unsigned short*)alloc((size_t)OUTF * HIDDEN * 2);
    float*          ws_xw1  = (float*)         alloc((size_t)N_FINE * HIDDEN * 4);
    float*          ws_out1 = (float*)         alloc((size_t)N_FINE * HIDDEN * 4);
    unsigned short* ws_h2bf = (unsigned short*)alloc((size_t)N_FINE * HIDDEN * 2);
    float*          ws_xw2  = (float*)         alloc((size_t)N_FINE * OUTF * 4);
    float*          ws_deg  = (float*)         alloc((size_t)N_FINE * 4);
    float*          ws_dis  = (float*)         alloc((size_t)N_FINE * 4);

    // 1) kNN (k=3) with LDS-staged coarse points
    knn_kernel<<<N_FINE / 256, 256, 0, stream>>>(pos, batch, pos_skip, batch_skip,
                                                 ws_idx, ws_w);
    // 2) weights -> bf16, transposed to [N,K]
    transpose_tobf16<<<(F_CAT * HIDDEN + 255) / 256, 256, 0, stream>>>(W1, ws_w1t, F_CAT, HIDDEN);
    transpose_tobf16<<<(HIDDEN * OUTF + 255) / 256, 256, 0, stream>>>(W2, ws_w2t, HIDDEN, OUTF);
    // 3) interpolate + concat -> bf16 activations
    interp_concat<<<(N_FINE * F_CAT) / 256, 256, 0, stream>>>(x, x_skip, ws_idx, ws_w, ws_hcat);
    // 4) GEMM1: [16384,768] x [768,512] via v_wmma_f32_16x16x32_bf16, 32x32/wave
    {
        int waves = (N_FINE / 32) * (HIDDEN / 32);   // 8192 waves, 8 waves/block
        wmma_gemm_bf16<F_CAT><<<waves / 8, 256, 0, stream>>>(ws_hcat, ws_w1t, ws_xw1,
                                                             N_FINE, HIDDEN);
    }
    // 5) shared degree normalization
    deg_init<<<N_FINE / 256, 256, 0, stream>>>(ws_deg);
    deg_count<<<N_EDGES / 256, 256, 0, stream>>>(edge_index, ws_deg);
    deg_rsqrt<<<N_FINE / 256, 256, 0, stream>>>(ws_deg, ws_dis);
    // 6) GCN layer 1 aggregate + bias + relu (-> bf16 for next GEMM)
    self_init<<<(N_FINE * HIDDEN) / 256, 256, 0, stream>>>(ws_xw1, ws_dis, ws_out1, HIDDEN);
    edge_scatter<<<N_EDGES / 8, 256, 0, stream>>>(edge_index, ws_dis, ws_xw1, ws_out1, HIDDEN);
    bias_relu_bf16<<<(N_FINE * HIDDEN) / 256, 256, 0, stream>>>(ws_out1, b1, ws_h2bf, HIDDEN);
    // 7) GEMM2: [16384,512] x [512,256]
    {
        int waves = (N_FINE / 32) * (OUTF / 32);     // 4096 waves
        wmma_gemm_bf16<HIDDEN><<<waves / 8, 256, 0, stream>>>(ws_h2bf, ws_w2t, ws_xw2,
                                                              N_FINE, OUTF);
    }
    // 8) GCN layer 2 aggregate directly into d_out, then bias+relu in place
    self_init<<<(N_FINE * OUTF) / 256, 256, 0, stream>>>(ws_xw2, ws_dis, outp, OUTF);
    edge_scatter<<<N_EDGES / 8, 256, 0, stream>>>(edge_index, ws_dis, ws_xw2, outp, OUTF);
    bias_relu<<<(N_FINE * OUTF) / 256, 256, 0, stream>>>(outp, b2, OUTF);
    // 9) tuple tail: pos_skip then batch_skip
    tail_copy<<<(N_FINE * 3 + 255) / 256, 256, 0, stream>>>(pos_skip, batch_skip,
                                                            outp + (size_t)N_FINE * OUTF);
}
